// MultiHeadAttention_70720931495942
// MI455X (gfx1250) — matrix operand order repaired and checked by reading, not compiled, not run on hardware
//
#include <hip/hip_runtime.h>

// ---- problem constants (match reference) ----
#define NHEAD   8
#define DK      64
#define DMODEL  512
#define BATCH   4
#define SLQ     2048
#define SLK     2048
#define NEGV    (-1000000000.0f)
#define INV_TEMP 0.125f   // 1/sqrt(64), folded into Wq

typedef __bf16 bf16;
typedef __attribute__((ext_vector_type(8)))  bf16  v8bf;
typedef __attribute__((ext_vector_type(16))) bf16  v16bf;
typedef __attribute__((ext_vector_type(8)))  float v8f;

#define WMMA_BF16(A, Bm, C) \
  __builtin_amdgcn_wmma_f32_16x16x32_bf16(false, (A), false, (Bm), (short)0, (C), false, false)

static __device__ __forceinline__ v8f vzero8() {
  v8f r;
#pragma unroll
  for (int i = 0; i < 8; ++i) r[i] = 0.0f;
  return r;
}

static __device__ __forceinline__ v16bf combine16(v8bf lo, v8bf hi) {
  v16bf r;
#pragma unroll
  for (int i = 0; i < 8; ++i) { r[i] = lo[i]; r[i + 8] = hi[i]; }
  return r;
}

// A-fragment (16x32 bf16, M x K) from row-major [16][ld] source.
// CDNA5 layout: lanes 0-15 -> M=lane, K {0..7,16..23}; lanes 16-31 -> M=lane-16, K {8..15,24..31}.
template <typename PT>
static __device__ __forceinline__ v16bf load_a_frag(PT base, int ld, int kb) {
  int lane = threadIdx.x & 31;
  int m    = lane & 15;
  int half = lane >> 4;
  const v8bf* p0 = (const v8bf*)(base + m * ld + kb + half * 8);
  const v8bf* p1 = (const v8bf*)(base + m * ld + kb + 16 + half * 8);
  return combine16(*p0, *p1);
}

// B-fragment (32x16 bf16, K x N) where column n is CONTIGUOUS over K in memory
// (i.e. source stored [N][K] row-major). Lanes 0-15: N=lane, K {0..7,16..23}; lanes 16-31: N=lane-16, K {8..15,24..31}.
template <typename PT>
static __device__ __forceinline__ v16bf load_bT_frag(PT base, int ld, int kb) {
  int lane = threadIdx.x & 31;
  int n    = lane & 15;
  int ks   = kb + (lane >> 4) * 8;
  const v8bf* p0 = (const v8bf*)(base + (size_t)n * ld + ks);
  const v8bf* p1 = (const v8bf*)(base + (size_t)n * ld + ks + 16);
  return combine16(*p0, *p1);
}

// ===================== kernel 1: per-head weight chains =====================
// Wq[h] = Wa*Wa*Wa2*Wa2 (scaled by 1/TEMP), stored transposed bf16 [h][n][k].
// Wv[h] = Wav*Wav, stored f32 [h][n][c].
__global__ void weights_kernel(const float* __restrict__ WA,
                               const float* __restrict__ WA2,
                               const float* __restrict__ WA3,
                               bf16* __restrict__ wqT,   // [H][64][64] (n-major)
                               float* __restrict__ wv) { // [H][64][64]
  __shared__ float sA[64 * 64];
  __shared__ float sB[64 * 64];
  __shared__ float sC[64 * 64];
  const int h = blockIdx.x;
  const int t = threadIdx.x; // 0..63, row index
  const float* A  = WA  + (size_t)h * 4096;
  const float* A2 = WA2 + (size_t)h * 4096;
  const float* A3 = WA3 + (size_t)h * 4096;

  for (int i = t; i < 4096; i += 64) sA[i] = A[i];
  __syncthreads();
  // T1 = A@A -> sB
  {
    float r[64];
#pragma unroll 4
    for (int j = 0; j < 64; ++j) {
      float s = 0.f;
      for (int k = 0; k < 64; ++k) s += sA[t * 64 + k] * sA[k * 64 + j];
      r[j] = s;
    }
    for (int j = 0; j < 64; ++j) sB[t * 64 + j] = r[j];
  }
  __syncthreads();
  for (int i = t; i < 4096; i += 64) sA[i] = A2[i];
  __syncthreads();
  // T2 = T1@A2 -> sC
  {
    float r[64];
#pragma unroll 4
    for (int j = 0; j < 64; ++j) {
      float s = 0.f;
      for (int k = 0; k < 64; ++k) s += sB[t * 64 + k] * sA[k * 64 + j];
      r[j] = s;
    }
    for (int j = 0; j < 64; ++j) sC[t * 64 + j] = r[j];
  }
  __syncthreads();
  // Wq = T2@A2, write transposed + scaled
  {
    for (int j = 0; j < 64; ++j) {
      float s = 0.f;
      for (int k = 0; k < 64; ++k) s += sC[t * 64 + k] * sA[k * 64 + j];
      wqT[(size_t)h * 4096 + j * 64 + t] = (bf16)(s * INV_TEMP);
    }
  }
  __syncthreads();
  for (int i = t; i < 4096; i += 64) sA[i] = A3[i];
  __syncthreads();
  // Wv = A3@A3
  for (int c = 0; c < 64; ++c) {
    float s = 0.f;
    for (int a = 0; a < 64; ++a) s += sA[t * 64 + a] * sA[a * 64 + c];
    wv[(size_t)h * 4096 + t * 64 + c] = s;
  }
}

// ===================== kernel 2: bf16 layout conversions =====================
__global__ void conv_k_kernel(const float* __restrict__ k, bf16* __restrict__ kbf) {
  // kbf[((b*H+h)*LK + l)*64 + d] = k[(b*LK + l)*512 + h*64 + d]
  size_t idx = (size_t)blockIdx.x * blockDim.x + threadIdx.x;
  size_t total = (size_t)BATCH * NHEAD * SLK * DK;
  if (idx >= total) return;
  int d = idx & 63;
  int l = (idx >> 6) & (SLK - 1);
  int h = (idx >> 17) & (NHEAD - 1);
  int b = idx >> 20;
  kbf[idx] = (bf16)k[((size_t)b * SLK + l) * DMODEL + h * DK + d];
}

__global__ void conv_vT_kernel(const float* __restrict__ v, bf16* __restrict__ vT) {
  // vT[((b*H+h)*64 + d)*LK + l] = v[(b*LK + l)*512 + h*64 + d]
  size_t idx = (size_t)blockIdx.x * blockDim.x + threadIdx.x;
  size_t total = (size_t)BATCH * NHEAD * DK * SLK;
  if (idx >= total) return;
  int l = idx & (SLK - 1);
  int d = (idx >> 11) & 63;
  int h = (idx >> 17) & (NHEAD - 1);
  int b = idx >> 20;
  vT[idx] = (bf16)v[((size_t)b * SLK + l) * DMODEL + h * DK + d];
}

__global__ void conv_fc_kernel(const float* __restrict__ fc, bf16* __restrict__ fcbf) {
  size_t idx = (size_t)blockIdx.x * blockDim.x + threadIdx.x;
  if (idx >= (size_t)DMODEL * DMODEL) return;
  fcbf[idx] = (bf16)fc[idx];
}

// ===================== kernel 3: QW = Q @ Wq (bf16 WMMA) =====================
// grid: BATCH*NHEAD*(LQ/128) blocks of 256 threads; wave w owns 16 rows, N=64, K=64.
__global__ void qw_kernel(const float* __restrict__ q,
                          const bf16* __restrict__ wqT,
                          bf16* __restrict__ qwbf) {
  __shared__ bf16 stage[8 * 16 * 64]; // per-wave Q tile, bf16
  const int qblk = blockIdx.x & 15;
  const int h    = (blockIdx.x >> 4) & (NHEAD - 1);
  const int b    = blockIdx.x >> 7;
  const int w    = threadIdx.x >> 5;
  const int lane = threadIdx.x & 31;
  const int half = lane >> 4;
  const int nlo  = lane & 15;
  const int q0   = qblk * 128 + w * 16;

  // stage Q[16][64] -> bf16 LDS (per-wave region; same-wave DS ops are in-order)
  const float* qsrc = q + ((size_t)b * SLQ + q0) * DMODEL + h * DK;
#pragma unroll
  for (int j = 0; j < 32; ++j) {
    int idx = lane * 32 + j;
    int r = idx >> 6, d = idx & 63;
    stage[w * 1024 + r * 64 + d] = (bf16)qsrc[(size_t)r * DMODEL + d];
  }
  asm volatile("s_wait_dscnt 0" ::: "memory");

  v16bf a0 = load_a_frag(stage + w * 1024, 64, 0);
  v16bf a1 = load_a_frag(stage + w * 1024, 64, 32);
  const bf16* wb = wqT + (size_t)h * 4096;

  v8f acc[4];
#pragma unroll
  for (int nt = 0; nt < 4; ++nt) {
    acc[nt] = vzero8();
    v16bf b0 = load_bT_frag(wb + nt * 16 * 64, 64, 0);
    v16bf b1 = load_bT_frag(wb + nt * 16 * 64, 64, 32);
    acc[nt] = WMMA_BF16(a0, b0, acc[nt]);
    acc[nt] = WMMA_BF16(a1, b1, acc[nt]);
  }

  bf16* out = qwbf + ((size_t)(b * NHEAD + h) * SLQ + q0) * DK;
#pragma unroll
  for (int nt = 0; nt < 4; ++nt)
#pragma unroll
    for (int i = 0; i < 8; ++i) {
      int M = i + 8 * half;
      out[(size_t)M * DK + nt * 16 + nlo] = (bf16)acc[nt][i];
    }
}

// ===================== kernel 4: fused attention =====================
// One block = (b, h, 16 q-rows). 8 waves; wave w owns a 256-wide column slice.
// S kept entirely in accumulators; softmax stats via shfl + LDS; normalized
// attn written once to HBM; P fed through LDS staging into P@V WMMAs.
__global__ void attn_kernel(const bf16* __restrict__ qwbf,
                            const bf16* __restrict__ kbf,
                            const bf16* __restrict__ vT,
                            const int*  __restrict__ mask,
                            const float* __restrict__ wv,
                            float* __restrict__ attnOut,
                            bf16* __restrict__ ctx) {
  __shared__ bf16  ldsP[8 * 16 * 32]; // per-wave P staging (16x32)
  __shared__ float Osh[16 * 64];
  __shared__ float rowmaxL[8 * 16];
  __shared__ float rowsumL[8 * 16];
  __shared__ float finalmax[16];
  __shared__ float finalinv[16];

  const int qt = blockIdx.x & 127;
  const int h  = (blockIdx.x >> 7) & (NHEAD - 1);
  const int b  = blockIdx.x >> 10;
  const int q0 = qt * 16;
  const int tid  = threadIdx.x;
  const int w    = tid >> 5;
  const int lane = tid & 31;
  const int half = lane >> 4;
  const int nlo  = lane & 15;

  for (int i = tid; i < 16 * 64; i += 256) Osh[i] = 0.0f;

  const bf16* qwb   = qwbf + ((size_t)(b * NHEAD + h) * SLQ + q0) * DK;
  const bf16* khead = kbf + (size_t)(b * NHEAD + h) * SLK * DK;
  const bf16* vhead = vT  + (size_t)(b * NHEAD + h) * DK * SLK;

  v16bf aq0 = load_a_frag(qwb, DK, 0);
  v16bf aq1 = load_a_frag(qwb, DK, 32);

  // ---- S = QW @ K^T over this wave's 256 columns (16 tiles) ----
  v8f acc[16];
#pragma unroll
  for (int t = 0; t < 16; ++t) {
    const bf16* kb = khead + (size_t)(w * 256 + t * 16) * DK;
    if (t < 15) __builtin_prefetch(khead + (size_t)(w * 256 + (t + 1) * 16) * DK, 0, 3);
    acc[t] = vzero8();
    v16bf b0 = load_bT_frag(kb, DK, 0);
    v16bf b1 = load_bT_frag(kb, DK, 32);
    acc[t] = WMMA_BF16(aq0, b0, acc[t]);
    acc[t] = WMMA_BF16(aq1, b1, acc[t]);
  }

  // ---- mask + per-row max ----
  float maxv[8];
#pragma unroll
  for (int i = 0; i < 8; ++i) maxv[i] = NEGV;
#pragma unroll
  for (int t = 0; t < 16; ++t) {
#pragma unroll
    for (int i = 0; i < 8; ++i) {
      int M = i + 8 * half;
      int N = w * 256 + t * 16 + nlo;
      int mv = mask[((size_t)b * SLQ + q0 + M) * SLK + N];
      float s = (mv != 0) ? acc[t][i] : NEGV;
      acc[t][i] = s;
      maxv[i] = fmaxf(maxv[i], s);
    }
  }
#pragma unroll
  for (int i = 0; i < 8; ++i) {
    maxv[i] = fmaxf(maxv[i], __shfl_xor(maxv[i], 1, 32));
    maxv[i] = fmaxf(maxv[i], __shfl_xor(maxv[i], 2, 32));
    maxv[i] = fmaxf(maxv[i], __shfl_xor(maxv[i], 4, 32));
    maxv[i] = fmaxf(maxv[i], __shfl_xor(maxv[i], 8, 32));
  }
  if (nlo == 0) {
#pragma unroll
    for (int i = 0; i < 8; ++i) rowmaxL[w * 16 + half * 8 + i] = maxv[i];
  }
  __syncthreads();
  if (tid < 16) {
    float fm = NEGV;
    for (int ww = 0; ww < 8; ++ww) fm = fmaxf(fm, rowmaxL[ww * 16 + tid]);
    finalmax[tid] = fm;
  }
  __syncthreads();

  // ---- exp + per-row sum ----
  float fm[8], sumv[8];
#pragma unroll
  for (int i = 0; i < 8; ++i) { fm[i] = finalmax[i + 8 * half]; sumv[i] = 0.0f; }
#pragma unroll
  for (int t = 0; t < 16; ++t) {
#pragma unroll
    for (int i = 0; i < 8; ++i) {
      float e = __expf(acc[t][i] - fm[i]);
      acc[t][i] = e;
      sumv[i] += e;
    }
  }
#pragma unroll
  for (int i = 0; i < 8; ++i) {
    sumv[i] += __shfl_xor(sumv[i], 1, 32);
    sumv[i] += __shfl_xor(sumv[i], 2, 32);
    sumv[i] += __shfl_xor(sumv[i], 4, 32);
    sumv[i] += __shfl_xor(sumv[i], 8, 32);
  }
  if (nlo == 0) {
#pragma unroll
    for (int i = 0; i < 8; ++i) rowsumL[w * 16 + half * 8 + i] = sumv[i];
  }
  __syncthreads();
  if (tid < 16) {
    float s = 0.0f;
    for (int ww = 0; ww < 8; ++ww) s += rowsumL[ww * 16 + tid];
    finalinv[tid] = 1.0f / s;
  }
  __syncthreads();
  float inv[8];
#pragma unroll
  for (int i = 0; i < 8; ++i) inv[i] = finalinv[i + 8 * half];

  // ---- write attn, stage P, and O += P @ V ----
  float* arow = attnOut + ((size_t)(b * NHEAD + h) * SLQ + q0) * SLK;
  v8f oacc[4];
#pragma unroll
  for (int nt = 0; nt < 4; ++nt) oacc[nt] = vzero8();

#pragma unroll
  for (int s = 0; s < 8; ++s) {
#pragma unroll
    for (int t2 = 0; t2 < 2; ++t2) {
      int t = 2 * s + t2;
#pragma unroll
      for (int i = 0; i < 8; ++i) {
        int M = i + 8 * half;
        int N = w * 256 + t * 16 + nlo;
        float p = acc[t][i] * inv[i];
        arow[(size_t)M * SLK + N] = p;
        ldsP[w * 512 + M * 32 + t2 * 16 + nlo] = (bf16)p;
      }
    }
    asm volatile("s_wait_dscnt 0" ::: "memory");
    v16bf ap = load_a_frag(ldsP + w * 512, 32, 0);
    int k0 = w * 256 + s * 32;
#pragma unroll
    for (int nt = 0; nt < 4; ++nt) {
      v16bf bv = load_bT_frag(vhead + (size_t)(nt * 16) * SLK, SLK, k0);
      oacc[nt] = WMMA_BF16(ap, bv, oacc[nt]);
    }
  }

  // reduce partial O across waves
#pragma unroll
  for (int nt = 0; nt < 4; ++nt)
#pragma unroll
    for (int i = 0; i < 8; ++i) {
      int M = i + 8 * half;
      atomicAdd(&Osh[M * 64 + nt * 16 + nlo], oacc[nt][i]);
    }
  __syncthreads();

  // ---- O2 = O @ Wv[h], write ctx bf16 [b, l, h*64+c] ----
  const float* wvh = wv + (size_t)h * 4096;
#pragma unroll
  for (int j = 0; j < 4; ++j) {
    int idx = tid * 4 + j;
    int m = idx >> 6, c = idx & 63;
    float s = 0.0f;
    for (int n = 0; n < 64; ++n) s += Osh[m * 64 + n] * wvh[n * 64 + c];
    ctx[((size_t)b * SLQ + q0 + m) * DMODEL + h * DK + c] = (bf16)s;
  }
}

// ===================== kernel 5: fc + residual + LayerNorm =====================
// Rows flattened [B*LQ, 512]. One block = 16 rows; wave w owns 64 output cols.
__global__ void fc_ln_kernel(const bf16* __restrict__ ctx,
                             const bf16* __restrict__ fcbf,  // [d][e]: row d contiguous over e(=K)
                             const float* __restrict__ qres,
                             const float* __restrict__ gamma,
                             const float* __restrict__ beta,
                             float* __restrict__ out) {
  __shared__ float ysh[16 * 512];
  __shared__ float redS[16 * 16];
  __shared__ float redQ[16 * 16];
  __shared__ float muL[16];
  __shared__ float rsL[16];

  const int m0   = blockIdx.x * 16;
  const int tid  = threadIdx.x;
  const int w    = tid >> 5;
  const int lane = tid & 31;
  const int half = lane >> 4;
  const int nlo  = lane & 15;
  const int nb   = w * 64;

  v8f acc[4];
#pragma unroll
  for (int nt = 0; nt < 4; ++nt) acc[nt] = vzero8();

  const bf16* abase = ctx + (size_t)m0 * DMODEL;
#pragma unroll 4
  for (int kc = 0; kc < 16; ++kc) {
    v16bf a = load_a_frag(abase, DMODEL, kc * 32);
#pragma unroll
    for (int nt = 0; nt < 4; ++nt) {
      v16bf bb = load_bT_frag(fcbf + (size_t)(nb + nt * 16) * DMODEL, DMODEL, kc * 32);
      acc[nt] = WMMA_BF16(a, bb, acc[nt]);
    }
  }

  // residual add, park in LDS
#pragma unroll
  for (int nt = 0; nt < 4; ++nt)
#pragma unroll
    for (int i = 0; i < 8; ++i) {
      int M = i + 8 * half;
      int N = nb + nt * 16 + nlo;
      float y = acc[nt][i] + qres[(size_t)(m0 + M) * DMODEL + N];
      ysh[M * 512 + N] = y;
    }
  __syncthreads();

  // LayerNorm: 16 threads per row
  {
    int row = tid >> 4;
    int t   = tid & 15;
    float s = 0.0f, ss = 0.0f;
    for (int e = t; e < 512; e += 16) {
      float v = ysh[row * 512 + e];
      s += v; ss += v * v;
    }
    redS[row * 16 + t] = s;
    redQ[row * 16 + t] = ss;
    __syncthreads();
    if (t == 0) {
      float ts = 0.0f, tq = 0.0f;
      for (int j = 0; j < 16; ++j) { ts += redS[row * 16 + j]; tq += redQ[row * 16 + j]; }
      float mu  = ts / 512.0f;
      float var = tq / 512.0f - mu * mu;
      muL[row] = mu;
      rsL[row] = rsqrtf(var + 1e-6f);
    }
    __syncthreads();
    float mu = muL[row], rs = rsL[row];
    for (int e = t; e < 512; e += 16) {
      float v = (ysh[row * 512 + e] - mu) * rs;
      out[(size_t)(m0 + row) * DMODEL + e] = v * gamma[e] + beta[e];
    }
  }
}

// ===================== launcher =====================
extern "C" void kernel_launch(void* const* d_in, const int* in_sizes, int n_in,
                              void* d_out, int out_size, void* d_ws, size_t ws_size,
                              hipStream_t stream) {
  const float* q     = (const float*)d_in[0];
  const float* k     = (const float*)d_in[1];
  const float* v     = (const float*)d_in[2];
  const int*   mask  = (const int*)d_in[3];
  const float* W_A   = (const float*)d_in[4];
  const float* W_A2  = (const float*)d_in[5];
  const float* W_A3  = (const float*)d_in[6];
  const float* fc_w  = (const float*)d_in[7];
  const float* gamma = (const float*)d_in[8];
  const float* beta  = (const float*)d_in[9];

  float* out_f   = (float*)d_out;                                     // [B,LQ,512]
  float* attnOut = out_f + (size_t)BATCH * SLQ * DMODEL;              // [B,H,LQ,LK]

  char* ws = (char*)d_ws;
  size_t off = 0;
  bf16*  wqT  = (bf16*)(ws + off);  off += (size_t)NHEAD * 64 * 64 * sizeof(bf16);
  float* wv   = (float*)(ws + off); off += (size_t)NHEAD * 64 * 64 * sizeof(float);
  bf16*  kbf  = (bf16*)(ws + off);  off += (size_t)BATCH * NHEAD * SLK * DK * sizeof(bf16);
  bf16*  vT   = (bf16*)(ws + off);  off += (size_t)BATCH * NHEAD * DK * SLK * sizeof(bf16);
  bf16*  qwbf = (bf16*)(ws + off);  off += (size_t)BATCH * NHEAD * SLQ * DK * sizeof(bf16);
  bf16*  fcbf = (bf16*)(ws + off);  off += (size_t)DMODEL * DMODEL * sizeof(bf16);
  bf16*  ctx  = (bf16*)(ws + off);  off += (size_t)BATCH * SLQ * DMODEL * sizeof(bf16);

  weights_kernel<<<NHEAD, 64, 0, stream>>>(W_A, W_A2, W_A3, wqT, wv);

  {
    size_t n = (size_t)BATCH * NHEAD * SLK * DK;
    int blocks = (int)((n + 255) / 256);
    conv_k_kernel<<<blocks, 256, 0, stream>>>(k, kbf);
    conv_vT_kernel<<<blocks, 256, 0, stream>>>(v, vT);
  }
  conv_fc_kernel<<<(DMODEL * DMODEL + 255) / 256, 256, 0, stream>>>(fc_w, fcbf);

  qw_kernel<<<BATCH * NHEAD * (SLQ / 128), 256, 0, stream>>>(q, wqT, qwbf);

  attn_kernel<<<BATCH * NHEAD * (SLQ / 16), 256, 0, stream>>>(
      qwbf, kbf, vT, mask, wv, attnOut, ctx);

  fc_ln_kernel<<<(BATCH * SLQ) / 16, 256, 0, stream>>>(ctx, fcbf, q, gamma, beta, out_f);
}
